// FourierBlock_74912819577314
// MI455X (gfx1250) — matrix operand order repaired and checked by reading, not compiled, not run on hardware
//
#include <hip/hip_runtime.h>
#include <hip/hip_bf16.h>

typedef _Float16 v16h __attribute__((ext_vector_type(16)));
typedef _Float16 v8h  __attribute__((ext_vector_type(8)));
typedef _Float16 v4h  __attribute__((ext_vector_type(4)));
typedef float    v8f  __attribute__((ext_vector_type(8)));

#define NB 4
#define NL 4096
#define NH 8
#define NE 64
#define NM 64
#define HE 512   // H*E rows
#define M2 128   // 2*M columns (re | im)
#define KSPLIT 4

// ws layout (hipMalloc base alignment >= 256B):
// [ 0, 1MB)   fwT : f16 [128][4096]   fwT[mm][l] = mm<64 ? cos(2pi m l/L) : -sin(...)
// [ 1, 2MB)   gT  : f16 [4096][128]   gT[l][mm]  = c_m * (mm<64 ? cos : sin), c_0=1/L else 2/L
// [ 2,18MB)   qT  : f16 [4][512][4096] transposed+converted q
// [18,22MB)   C1p : f32 [4 ksplit][4][512][128] forward DFT partials
// [22,22.5MB) RmH : f16 [4][512][128] head-mixed coefficients

__global__ __launch_bounds__(128) void k_twiddle(_Float16* __restrict__ fwT,
                                                 _Float16* __restrict__ gT) {
    const int l  = blockIdx.x;          // 0..4095
    const int mm = threadIdx.x;         // 0..127
    const int m  = mm & 63;
    const int phase = (m * l) & (NL - 1);   // exact integer phase reduction
    float s, c;
    __sincosf((float)phase * (6.28318530717958647692f / (float)NL), &s, &c);
    const float cm = (m == 0) ? (1.0f / NL) : (2.0f / NL);   // irfft scale folded into gT
    fwT[(size_t)mm * NL + l] = (_Float16)((mm < 64) ? c : -s);
    gT[(size_t)l * M2 + mm]  = (_Float16)(cm * ((mm < 64) ? c : s));
}

// qT[b][he][l] = (f16) q[b][l][he]   -- 64x64 tiles through LDS
__global__ __launch_bounds__(256) void k_transpose(const float* __restrict__ q,
                                                   _Float16* __restrict__ qT) {
    __shared__ float tile[64 * 68];
    const int blk = blockIdx.x;              // 2048: b(4) x lt(64) x het(8)
    const int b   = blk >> 9;
    const int rem = blk & 511;
    const int lt  = (rem >> 3) << 6;
    const int het = (rem & 7) << 6;
    const int t   = threadIdx.x;
    #pragma unroll
    for (int p = 0; p < 4; ++p) {            // coalesced f32 reads: rows = l
        const int i   = p * 16 + (t >> 4);
        const int seg = (t & 15) << 2;
        const float4 v = *(const float4*)&q[((size_t)b * NL + lt + i) * HE + het + seg];
        *(float4*)&tile[i * 68 + seg] = v;
    }
    __syncthreads();
    #pragma unroll
    for (int p = 0; p < 4; ++p) {            // coalesced f16 writes: rows = he
        const int jj = p * 16 + (t >> 4);
        const int ii = (t & 15) << 2;
        v4h hv;
        hv.x = (_Float16)tile[(ii + 0) * 68 + jj];
        hv.y = (_Float16)tile[(ii + 1) * 68 + jj];
        hv.z = (_Float16)tile[(ii + 2) * 68 + jj];
        hv.w = (_Float16)tile[(ii + 3) * 68 + jj];
        *(v4h*)&qT[((size_t)b * HE + het + jj) * NL + lt + ii] = hv;
    }
}

// C1p[ks][b][he][mm] = sum_{l in ks chunk} qT[b][he][l] * fwT[mm][l]
// per wave: 16(he) x 32(mm) tile, K=1024 chunk, pure global b128 loads + WMMA
__global__ __launch_bounds__(128) void k_dft_fwd(const _Float16* __restrict__ qT,
                                                 const _Float16* __restrict__ fwT,
                                                 float* __restrict__ C1p) {
    const int blk    = blockIdx.x;        // 512: ks(4) x b(4) x hetile(32); wave -> mm pair
    const int ks     = blk >> 7;
    const int b      = (blk >> 5) & 3;
    const int hetile = (blk & 31) << 4;
    const int wave   = threadIdx.x >> 5, lane = threadIdx.x & 31;
    const int mmt0   = wave << 5;         // 4 waves cover mm = 0..127
    const int mmt1   = mmt0 + 16;
    const int mrow   = lane & 15;
    const int kb8    = (lane >> 4) << 3;
    const int kb16   = (lane >> 4) << 4;
    const int ncol   = lane & 15;

    const _Float16* arow = &qT[((size_t)b * HE + hetile + mrow) * NL + kb16];
    const _Float16* br0  = &fwT[(size_t)(mmt0 + ncol) * NL + kb16];
    const _Float16* br1  = &fwT[(size_t)(mmt1 + ncol) * NL + kb16];
    const int k0 = ks << 10, k1 = k0 + 1024;

    v8f acc0 = {}, acc1 = {};
    for (int kt = k0; kt < k1; kt += 32) {
        const v8h c0 = *(const v8h*)&arow[kt];        // A halves: K = base..base+7
        const v8h c1 = *(const v8h*)&arow[kt + 8];    //           K = base+8..base+15
        const v16h a = __builtin_shufflevector(c0, c1, 0,1,2,3,4,5,6,7,
                                                       8,9,10,11,12,13,14,15);
        const v16h b0 = *(const v16h*)&br0[kt];
        const v16h b1 = *(const v16h*)&br1[kt];
        acc0 = __builtin_amdgcn_wmma_f32_16x16x32_f16(false, a, false, b0,
                                                      (short)0, acc0, false, false);
        acc1 = __builtin_amdgcn_wmma_f32_16x16x32_f16(false, a, false, b1,
                                                      (short)0, acc1, false, false);
    }
    float* crow = &C1p[(((size_t)ks * NB + b) * HE + hetile + kb8) * M2 + ncol];
    #pragma unroll
    for (int r = 0; r < 8; ++r) {
        crow[(size_t)r * M2 + mmt0] = acc0[r];
        crow[(size_t)r * M2 + mmt1] = acc1[r];
    }
}

// res[b,o,e,m] = sum_i X[b,i,e,m]*W[i,o,e,m]  (X = sum of K-split partials), f16 out
__global__ __launch_bounds__(256) void k_mix(const float* __restrict__ C1p,
                                             const float* __restrict__ wr,
                                             const float* __restrict__ wi,
                                             _Float16* __restrict__ RmH) {
    const int t = blockIdx.x * 256 + threadIdx.x;   // 131072 = B*H*E*M
    const int m = t & 63;
    const int e = (t >> 6) & 63;
    const int o = (t >> 12) & 7;
    const int b = t >> 15;
    float rr = 0.f, ri = 0.f;
    #pragma unroll
    for (int i = 0; i < NH; ++i) {
        float xr = 0.f, xi = 0.f;
        #pragma unroll
        for (int ks = 0; ks < KSPLIT; ++ks) {
            const size_t xo = (((size_t)ks * NB + b) * HE + i * NE + e) * M2 + m;
            xr += C1p[xo];
            xi += C1p[xo + 64];
        }
        const size_t wo = (((size_t)i * NH + o) * NE + e) * NM + m;
        const float a = wr[wo], bb = wi[wo];
        rr += xr * a - xi * bb;
        ri += xr * bb + xi * a;
    }
    const size_t ro = ((size_t)b * HE + o * NE + e) * M2;
    RmH[ro + m]      = (_Float16)rr;    // cos coefficient (scale lives in gT)
    RmH[ro + 64 + m] = (_Float16)(-ri); // sin coefficient
}

// out[b][oe][l] = sum_mm RmH[b][oe][mm] * G[mm][l]   per wave: 16(oe) x 32(l), K=128
__global__ __launch_bounds__(128) void k_dft_inv(const _Float16* __restrict__ RmH,
                                                 const _Float16* __restrict__ gT,
                                                 float* __restrict__ out) {
    const int blk    = blockIdx.x;        // 4096: b(4) x oetile(32) x lgrp(32); wave -> l pair
    const int b      = blk >> 10;
    const int rem    = blk & 1023;
    const int oetile = (rem >> 5) << 4;
    const int lgrp   = rem & 31;
    const int wave   = threadIdx.x >> 5, lane = threadIdx.x & 31;
    const int lt0    = ((lgrp << 2) + wave) << 5;   // 128 l-pairs per batch row-strip
    const int lt1    = lt0 + 16;
    const int mrow   = lane & 15;
    const int kb8    = (lane >> 4) << 3;
    const int kb16   = (lane >> 4) << 4;
    const int ncol   = lane & 15;

    const _Float16* arow = &RmH[((size_t)b * HE + oetile + mrow) * M2 + kb16];
    const _Float16* br0  = &gT[(size_t)(lt0 + ncol) * M2 + kb16];
    const _Float16* br1  = &gT[(size_t)(lt1 + ncol) * M2 + kb16];

    v8f acc0 = {}, acc1 = {};
    #pragma unroll
    for (int kt = 0; kt < M2; kt += 32) {
        const v8h c0 = *(const v8h*)&arow[kt];
        const v8h c1 = *(const v8h*)&arow[kt + 8];
        const v16h a = __builtin_shufflevector(c0, c1, 0,1,2,3,4,5,6,7,
                                                       8,9,10,11,12,13,14,15);
        const v16h b0 = *(const v16h*)&br0[kt];
        const v16h b1 = *(const v16h*)&br1[kt];
        acc0 = __builtin_amdgcn_wmma_f32_16x16x32_f16(false, a, false, b0,
                                                      (short)0, acc0, false, false);
        acc1 = __builtin_amdgcn_wmma_f32_16x16x32_f16(false, a, false, b1,
                                                      (short)0, acc1, false, false);
    }
    float* orow = &out[((size_t)b * HE + oetile + kb8) * NL + ncol];
    #pragma unroll
    for (int r = 0; r < 8; ++r) {         // 64B-coalesced row segments
        orow[(size_t)r * NL + lt0] = acc0[r];
        orow[(size_t)r * NL + lt1] = acc1[r];
    }
}

extern "C" void kernel_launch(void* const* d_in, const int* in_sizes, int n_in,
                              void* d_out, int out_size, void* d_ws, size_t ws_size,
                              hipStream_t stream) {
    (void)in_sizes; (void)n_in; (void)out_size; (void)ws_size;
    const float* q  = (const float*)d_in[0];   // [B,L,H,E]
    const float* wr = (const float*)d_in[3];   // [H,H,E,M]
    const float* wi = (const float*)d_in[4];   // [H,H,E,M]
    float* out = (float*)d_out;                // [B,H,E,L]

    char* ws = (char*)d_ws;
    _Float16* fwT = (_Float16*)(ws);
    _Float16* gT  = (_Float16*)(ws + (size_t)(1u << 20));
    _Float16* qT  = (_Float16*)(ws + (size_t)(2u << 20));
    float*    C1p = (float*)   (ws + (size_t)(18u << 20));
    _Float16* RmH = (_Float16*)(ws + (size_t)(22u << 20));

    k_twiddle  <<<NL,   128, 0, stream>>>(fwT, gT);
    k_transpose<<<2048, 256, 0, stream>>>(q, qT);
    k_dft_fwd  <<<512,  128, 0, stream>>>(qT, fwT, C1p);
    k_mix      <<<512,  256, 0, stream>>>(C1p, wr, wi, RmH);
    k_dft_inv  <<<4096, 128, 0, stream>>>(RmH, gT, out);
}
